// FragmentedLinear_80075370267207
// MI455X (gfx1250) — matrix operand order repaired and checked by reading, not compiled, not run on hardware
//
#include <hip/hip_runtime.h>

// ---------------------------------------------------------------------------
// FragmentedLinear for MI455X (gfx1250): bf16 WMMA (f32 accum) pipeline.
//   K_init    : zero expert histogram
//   K_convert : f32 -> bf16 weight conversion (compressor_w, compressed_net_w)
//   K_transpT : expert_weights [e][k][n] f32 -> [e][n][k] bf16 (LDS transpose)
//   K_selector: per-token scores + argmax + x -> bf16, histogram
//   K_scan    : prefix-sum + scatter to build per-expert token permutation
//   K_gemm<1> : grouped expert GEMM   -> d_out (store)
//   K_gemm<0> : masked compressor GEMM-> c1 (bf16)
//   K_gemm<2> : compressed-net GEMM   -> d_out (accumulate)
// GEMM inner loop: double-buffered B tile in LDS filled with
// GLOBAL_LOAD_ASYNC_TO_LDS_B128 (ASYNCcnt), wave tile M32xN64 (8 WMMA/step).
// ---------------------------------------------------------------------------

#define BATCH   8192
#define INF     4096
#define OUTF    4096
#define NFRAG   16
#define FRAGSZ  256
#define COMP    512

typedef __bf16 bf16x16 __attribute__((ext_vector_type(16)));
typedef float  f32x8   __attribute__((ext_vector_type(8)));

union FragU { bf16x16 v; uint4 q[2]; };

__device__ __forceinline__ unsigned short f2bf(float f) {
    union { float f; unsigned u; } c; c.f = f;
    unsigned r = c.u + 0x7FFFu + ((c.u >> 16) & 1u);   // round-to-nearest-even
    return (unsigned short)(r >> 16);
}

// ---------------------------------------------------------------------------
__global__ void K_init(int* counts) {
    if (threadIdx.x < NFRAG) counts[threadIdx.x] = 0;
}

// f32 -> bf16 elementwise (n multiple of 1024)
__global__ __launch_bounds__(256) void K_convert(const float* __restrict__ in,
                                                 unsigned short* __restrict__ out) {
    int i = (blockIdx.x * 256 + threadIdx.x) * 4;
    float4 v = *(const float4*)(in + i);
    ushort4 o; o.x = f2bf(v.x); o.y = f2bf(v.y); o.z = f2bf(v.z); o.w = f2bf(v.w);
    *(ushort4*)(out + i) = o;
}

// expert_weights [16][256][4096] f32 -> [16][4096][256] bf16
__global__ __launch_bounds__(256) void K_transpT(const float* __restrict__ ew,
                                                 unsigned short* __restrict__ ewT) {
    __shared__ float t[32][33];
    int e = blockIdx.z, nb = blockIdx.x * 32, kb = blockIdx.y * 32;
    const float* src = ew + ((size_t)e * FRAGSZ + kb) * OUTF + nb;
#pragma unroll
    for (int j = 0; j < 32; j += 8)
        t[threadIdx.y + j][threadIdx.x] = src[(size_t)(threadIdx.y + j) * OUTF + threadIdx.x];
    __syncthreads();
    unsigned short* dst = ewT + ((size_t)e * OUTF + nb) * FRAGSZ + kb;
#pragma unroll
    for (int j = 0; j < 32; j += 8)
        dst[(size_t)(threadIdx.y + j) * FRAGSZ + threadIdx.x] = f2bf(t[threadIdx.x][threadIdx.y + j]);
}

// one wave per token: selector dots + argmax, x row -> bf16, expert histogram
__global__ __launch_bounds__(256) void K_selector(const float* __restrict__ x,
                                                  const float* __restrict__ selw,
                                                  unsigned short* __restrict__ xbf,
                                                  int* __restrict__ selOut,
                                                  int* __restrict__ counts) {
    int wave = threadIdx.x >> 5, lane = threadIdx.x & 31;
    int token = blockIdx.x * 8 + wave;
    const float* xr = x + (size_t)token * INF + lane * 128;           // 128 elems/lane
    const float* wr = selw + (lane >> 1) * FRAGSZ + (lane & 1) * 128; // lane covers half a fragment
    unsigned short* ob = xbf + (size_t)token * INF + lane * 128;
    float part = 0.f;
#pragma unroll 4
    for (int j = 0; j < 128; j += 4) {
        float4 xv = *(const float4*)(xr + j);
        float4 wv = *(const float4*)(wr + j);
        part += xv.x * wv.x + xv.y * wv.y + xv.z * wv.z + xv.w * wv.w;
        ushort4 o; o.x = f2bf(xv.x); o.y = f2bf(xv.y); o.z = f2bf(xv.z); o.w = f2bf(xv.w);
        *(ushort4*)(ob + j) = o;
    }
    part += __shfl_xor(part, 1);          // lanes 2f,2f+1 now hold score f
    float best = -1e30f; int bidx = 0;
#pragma unroll
    for (int f = 0; f < NFRAG; ++f) {
        float s = __shfl(part, 2 * f);
        if (s > best) { best = s; bidx = f; }   // strict > : first max, matches argmax
    }
    if (lane == 0) { selOut[token] = bidx; atomicAdd(&counts[bidx], 1); }
}

// single block: prefix-sum of counts + scatter token ids into per-expert groups
__global__ __launch_bounds__(256) void K_scan(const int* __restrict__ sel,
                                              const int* __restrict__ counts,
                                              int* __restrict__ offsets,
                                              int* __restrict__ perm) {
    __shared__ int cur[NFRAG];
    if (threadIdx.x == 0) {
        int a = 0;
        for (int f = 0; f < NFRAG; ++f) { offsets[f] = a; cur[f] = a; a += counts[f]; }
    }
    __syncthreads();
    for (int t = threadIdx.x; t < BATCH; t += 256) {
        int pos = atomicAdd(&cur[sel[t]], 1);
        perm[pos] = t;
    }
}

// ---------------------------------------------------------------------------
// Tiled bf16 GEMM: block = 256M x 64N, K-step 32, 8 waves (wave = 32 rows).
// A: [*,K] bf16 row-major (per-lane rows, gathered for MODE==1, masked MODE==0)
// B: [N,K] bf16 row-major, double-buffered 64x32 LDS tile filled with
//    global_load_async_to_lds_b128 (ASYNCcnt pipelined one tile ahead).
// MODE 0: compressor (mask selected fragment, store bf16 c1)
// MODE 1: expert     (gather rows via perm, store f32 to d_out, row guards)
// MODE 2: net        (accumulate f32 into d_out)
// ---------------------------------------------------------------------------
template <int MODE>
__global__ __launch_bounds__(256, 2)
void K_gemm(const unsigned short* __restrict__ A, const unsigned short* __restrict__ B,
            void* __restrict__ Cout, const int* __restrict__ sel,
            const int* __restrict__ perm, const int* __restrict__ counts,
            const int* __restrict__ offsets, int K) {
    __shared__ __align__(16) unsigned short ldsB[2][64 * 32];
    const int tid = threadIdx.x;
    const int wave = tid >> 5, lane = tid & 31;
    const int m = lane & 15, half = lane >> 4;

    const int nbase = blockIdx.x * 64;
    const int mtile = blockIdx.y;
    int cnt = 0, gstart = 0;
    if (MODE == 1) {
        int e = blockIdx.z;
        cnt = counts[e]; gstart = offsets[e];
        if (mtile * 256 >= cnt) return;
        B += (size_t)e * OUTF * FRAGSZ;
    }
    const int mbase = mtile * 256 + wave * 32;

    int rowIdx0, rowIdx1;
    if (MODE == 1) {
        int rl0 = mbase + m, rl1 = mbase + 16 + m;
        rowIdx0 = perm[gstart + ((rl0 < cnt) ? rl0 : 0)];
        rowIdx1 = perm[gstart + ((rl1 < cnt) ? rl1 : 0)];
    } else {
        rowIdx0 = mbase + m;
        rowIdx1 = mbase + 16 + m;
    }
    const unsigned short* Arow0 = A + (size_t)rowIdx0 * K;
    const unsigned short* Arow1 = A + (size_t)rowIdx1 * K;
    int sel0 = 0, sel1 = 0;
    if (MODE == 0) { sel0 = sel[rowIdx0]; sel1 = sel[rowIdx1]; }

    f32x8 acc[2][4];
#pragma unroll
    for (int mf = 0; mf < 2; ++mf)
#pragma unroll
        for (int nf = 0; nf < 4; ++nf)
#pragma unroll
            for (int j = 0; j < 8; ++j) acc[mf][nf][j] = 0.f;

    const int ksteps = K >> 5;
    const int bn = tid >> 2, bc = tid & 3;                 // B stager: 64 rows x 4 chunks
    const unsigned short* Bst = B + (size_t)(nbase + bn) * K + bc * 8;
    unsigned ldsDst0 = (unsigned)(size_t)&ldsB[0][bn * 32 + bc * 8];
    unsigned ldsDst1 = (unsigned)(size_t)&ldsB[1][bn * 32 + bc * 8];

    // prologue: async-stage tile 0 into buffer 0
    asm volatile("global_load_async_to_lds_b128 %0, %1, off"
                 :: "v"(ldsDst0), "v"(Bst) : "memory");

    for (int ks = 0; ks < ksteps; ++ks) {
        const int k0 = ks << 5;
        if (ks + 1 < ksteps) {
            unsigned dst = ((ks + 1) & 1) ? ldsDst1 : ldsDst0;
            asm volatile("global_load_async_to_lds_b128 %0, %1, off"
                         :: "v"(dst), "v"(Bst + k0 + 32) : "memory");
            asm volatile("s_wait_asynccnt 1" ::: "memory");  // tile ks landed, ks+1 in flight
        } else {
            asm volatile("s_wait_asynccnt 0" ::: "memory");
        }
        __syncthreads();

        // A fragments: 16-bit 16x32, lane m, halves carry K {h*8..h*8+7, 16+h*8..}
        FragU a0, a1;
        a0.q[0] = *(const uint4*)(Arow0 + k0 + half * 8);
        a0.q[1] = *(const uint4*)(Arow0 + k0 + 16 + half * 8);
        a1.q[0] = *(const uint4*)(Arow1 + k0 + half * 8);
        a1.q[1] = *(const uint4*)(Arow1 + k0 + 16 + half * 8);
        if (MODE == 0) {                                   // K-tile inside selected fragment?
            int fr = k0 >> 8;
            if (sel0 == fr) { a0.q[0] = make_uint4(0, 0, 0, 0); a0.q[1] = make_uint4(0, 0, 0, 0); }
            if (sel1 == fr) { a1.q[0] = make_uint4(0, 0, 0, 0); a1.q[1] = make_uint4(0, 0, 0, 0); }
        }
        const unsigned short* lb = ldsB[ks & 1];
#pragma unroll
        for (int nf = 0; nf < 4; ++nf) {
            FragU b;
            const unsigned short* bp = &lb[(nf * 16 + m) * 32 + half * 8];
            b.q[0] = *(const uint4*)(bp);
            b.q[1] = *(const uint4*)(bp + 16);
            acc[0][nf] = __builtin_amdgcn_wmma_f32_16x16x32_bf16(
                false, a0.v, false, b.v, (short)0, acc[0][nf], false, false);
            acc[1][nf] = __builtin_amdgcn_wmma_f32_16x16x32_bf16(
                false, a1.v, false, b.v, (short)0, acc[1][nf], false, false);
        }
        __syncthreads();   // buffer ks&1 free for reuse at iter ks+1's async issue
    }

    // C/D layout: lane -> N = lane&15 ; VGPR r -> M = r + 8*(lane>>4)
    if (MODE == 0) {
        unsigned short* C = (unsigned short*)Cout;          // c1 [BATCH, COMP] bf16
#pragma unroll
        for (int mf = 0; mf < 2; ++mf)
#pragma unroll
            for (int nf = 0; nf < 4; ++nf) {
                int col = nbase + nf * 16 + m;
#pragma unroll
                for (int r = 0; r < 8; ++r)
                    C[(size_t)(mbase + mf * 16 + r + 8 * half) * COMP + col] = f2bf(acc[mf][nf][r]);
            }
    } else if (MODE == 1) {
        float* C = (float*)Cout;
#pragma unroll
        for (int mf = 0; mf < 2; ++mf) {
            int ri = mf ? rowIdx1 : rowIdx0;
#pragma unroll
            for (int r = 0; r < 8; ++r) {
                int rl = mbase + mf * 16 + r + 8 * half;
                int row = __shfl(ri, r + 8 * half);
                if (rl < cnt) {
#pragma unroll
                    for (int nf = 0; nf < 4; ++nf)
                        C[(size_t)row * OUTF + nbase + nf * 16 + m] = acc[mf][nf][r];
                }
            }
        }
    } else {
        float* C = (float*)Cout;
#pragma unroll
        for (int mf = 0; mf < 2; ++mf)
#pragma unroll
            for (int nf = 0; nf < 4; ++nf) {
                int col = nbase + nf * 16 + m;
#pragma unroll
                for (int r = 0; r < 8; ++r) {
                    size_t idx = (size_t)(mbase + mf * 16 + r + 8 * half) * OUTF + col;
                    C[idx] += acc[mf][nf][r];
                }
            }
    }
}

// ---------------------------------------------------------------------------
extern "C" void kernel_launch(void* const* d_in, const int* in_sizes, int n_in,
                              void* d_out, int out_size, void* d_ws, size_t ws_size,
                              hipStream_t stream) {
    const float* x    = (const float*)d_in[0];
    const float* selw = (const float*)d_in[1];
    const float* ew   = (const float*)d_in[2];
    const float* wc   = (const float*)d_in[3];
    const float* wn   = (const float*)d_in[4];

    char* ws = (char*)d_ws;
    size_t off = 0;
    auto alloc = [&](size_t bytes) -> void* {
        void* p = ws + off; off += (bytes + 255) & ~(size_t)255; return p;
    };
    unsigned short* xbf = (unsigned short*)alloc((size_t)BATCH * INF * 2);        // 64 MiB
    unsigned short* ewT = (unsigned short*)alloc((size_t)16 * OUTF * FRAGSZ * 2); // 32 MiB
    unsigned short* wcb = (unsigned short*)alloc((size_t)COMP * INF * 2);         //  4 MiB
    unsigned short* wnb = (unsigned short*)alloc((size_t)OUTF * COMP * 2);        //  4 MiB
    unsigned short* c1  = (unsigned short*)alloc((size_t)BATCH * COMP * 2);       //  8 MiB
    int* sel     = (int*)alloc(BATCH * 4);
    int* perm    = (int*)alloc(BATCH * 4);
    int* counts  = (int*)alloc(64);
    int* offsets = (int*)alloc(64);

    K_init<<<1, 32, 0, stream>>>(counts);
    K_convert<<<(COMP * INF / 4) / 256, 256, 0, stream>>>(wc, wcb);
    K_convert<<<(OUTF * COMP / 4) / 256, 256, 0, stream>>>(wn, wnb);
    K_transpT<<<dim3(OUTF / 32, FRAGSZ / 32, 16), dim3(32, 8), 0, stream>>>(ew, ewT);
    K_selector<<<BATCH / 8, 256, 0, stream>>>(x, selw, xbf, sel, counts);
    K_scan<<<1, 256, 0, stream>>>(sel, counts, offsets, perm);

    // expert grouped GEMM: overwrites every row of d_out
    K_gemm<1><<<dim3(OUTF / 64, BATCH / 256, 16), 256, 0, stream>>>(
        xbf, ewT, d_out, sel, perm, counts, offsets, FRAGSZ);
    // masked compressor GEMM -> c1 (bf16)
    K_gemm<0><<<dim3(COMP / 64, BATCH / 256, 1), 256, 0, stream>>>(
        xbf, wcb, c1, sel, perm, counts, offsets, INF);
    // compressed-net GEMM: accumulates into d_out
    K_gemm<2><<<dim3(OUTF / 64, BATCH / 256, 1), 256, 0, stream>>>(
        c1, wnb, d_out, sel, perm, counts, offsets, COMP);
}
